// DecLayer_84954453115393
// MI455X (gfx1250) — matrix operand order repaired and checked by reading, not compiled, not run on hardware
//
#include <hip/hip_runtime.h>

typedef __bf16 v16bf __attribute__((ext_vector_type(16)));
typedef float  v8f   __attribute__((ext_vector_type(8)));
typedef unsigned int v4u __attribute__((ext_vector_type(4)));

struct BF16Frag { v4u lo, hi; };

// ---- helpers -------------------------------------------------------------

__device__ __forceinline__ unsigned short f2bf(float f) {
  unsigned int u = __builtin_bit_cast(unsigned int, f);
  u += 0x7FFFu + ((u >> 16) & 1u);            // round-to-nearest-even
  return (unsigned short)(u >> 16);
}
__device__ __forceinline__ unsigned int pack2(float a, float b) {
  return (unsigned int)f2bf(a) | ((unsigned int)f2bf(b) << 16);
}
__device__ __forceinline__ float gelu(float x) {
  return 0.5f * x * (1.0f + erff(x * 0.70710678118654752f));
}
// Weight (B) fragment: 16 contiguous bf16 per lane (packed by pack_weights).
__device__ __forceinline__ v16bf load_bfrag(const unsigned short* p) {
  BF16Frag s; s.lo = *(const v4u*)p; s.hi = *(const v4u*)(p + 8);
  return __builtin_bit_cast(v16bf, s);
}
// A fragment from row-major bf16 LDS row: ISA layout -> two 8-element runs,
// e=0..7 at k0, e=8..15 at k0+16 (k0 already includes +8*half).
__device__ __forceinline__ v16bf load_afrag(const unsigned short* p) {
  BF16Frag s; s.lo = *(const v4u*)p; s.hi = *(const v4u*)(p + 16);
  return __builtin_bit_cast(v16bf, s);
}
__device__ __forceinline__ v8f wmma_bf16(v16bf a, v16bf b, v8f c) {
  return __builtin_amdgcn_wmma_f32_16x16x32_bf16(
      /*neg_a=*/false, a, /*neg_b=*/false, b,
      /*c_mod=*/(short)0, c, /*reuse_a=*/false, /*reuse_b=*/false);
}

// ---- weight packing: f32 row-major [Kdim,Ndim] -> bf16 B-fragment order ---
// dst index d = ((nt*ksteps + ks)*32 + lane)*16 + e
// element   = W[ ks*32 + 8*(lane/16) + (e&7) + 16*(e/8) , nt*16 + lane%16 ]
__global__ void pack_weights_kernel(const float* __restrict__ src,
                                    unsigned short* __restrict__ dst,
                                    int Kdim, int Ndim) {
  int d = blockIdx.x * blockDim.x + threadIdx.x;
  int total = Kdim * Ndim;
  if (d >= total) return;
  int e    = d & 15;
  int lane = (d >> 4) & 31;
  int ts   = d >> 9;
  int ksteps = Kdim >> 5;
  int ks = ts % ksteps;
  int nt = ts / ksteps;
  int k = ks * 32 + ((lane >> 4) << 3) + (e & 7) + ((e >> 3) << 4);
  int n = (nt << 4) + (lane & 15);
  dst[d] = f2bf(src[k * Ndim + n]);
}

// ---- main layer kernel ----------------------------------------------------
#define HC 128
#define SC 392            // h_cat LDS row stride (ushorts), padded: bank-clean
#define SM 136            // mid row stride
#define SF 264            // ffn-hidden row stride
// workspace offsets (ushort units)
#define OW1   0
#define OW2   49152
#define OW3   65536
#define OWIN  81920
#define OWOUT 114688
// LDS arena (bytes), phase-aliased
#define L_HCAT 0          // node loop: [32][392] bf16  (25088 B)
#define L_LNF  0          // ffn phase: [16][128] f32   ( 8192 B)
#define L_LNB  8192       // ffn phase: [16][136] bf16  ( 4352 B)
#define L_FF   12544      // ffn phase: [16][264] bf16  ( 8448 B)
#define L_MID  25088      // node loop: [32][136] bf16  ( 8704 B)
#define L_FO   25088      // ffn phase: [16][128] f32   ( 8192 B)
#define L_MID2 33792      // node loop: [32][136] bf16  ( 8704 B)
#define L_DH   42496      // [16][128] f32              ( 8192 B)
#define LDS_BYTES 50688

__global__ __launch_bounds__(256) void declayer_kernel(
    const float* __restrict__ hV,   const float* __restrict__ hEV,
    const float* __restrict__ W1b,  const float* __restrict__ W2b,
    const float* __restrict__ W3b,
    const float* __restrict__ ln1g, const float* __restrict__ ln1b,
    const float* __restrict__ ln2g, const float* __restrict__ ln2b,
    const float* __restrict__ Winb, const float* __restrict__ Woutb,
    const unsigned short* __restrict__ wsw, float* __restrict__ out) {

  __shared__ __align__(16) char smem[LDS_BYTES];
  unsigned short* s_hcat = (unsigned short*)(smem + L_HCAT);
  unsigned short* s_mid  = (unsigned short*)(smem + L_MID);
  unsigned short* s_mid2 = (unsigned short*)(smem + L_MID2);
  float*          s_dh   = (float*)(smem + L_DH);
  float*          s_lnf  = (float*)(smem + L_LNF);
  unsigned short* s_lnb  = (unsigned short*)(smem + L_LNB);
  unsigned short* s_ff   = (unsigned short*)(smem + L_FF);
  float*          s_fo   = (float*)(smem + L_FO);

  const int tid  = threadIdx.x;
  const int wv   = tid >> 5;          // wave 0..7 -> owns N-tile `wv`
  const int lane = tid & 31;
  const int hf   = lane >> 4;         // half-wave
  const int lm   = lane & 15;
  const int nodeBase = blockIdx.x * 16;

  // Persistent weight fragments for this wave's N-tile (L2-hot loads)
  v16bf w1f[12], w2f[4], w3f[4];
#pragma unroll
  for (int ks = 0; ks < 12; ++ks)
    w1f[ks] = load_bfrag(wsw + OW1 + ((wv * 12 + ks) * 32 + lane) * 16);
#pragma unroll
  for (int ks = 0; ks < 4; ++ks)
    w2f[ks] = load_bfrag(wsw + OW2 + ((wv * 4 + ks) * 32 + lane) * 16);
#pragma unroll
  for (int ks = 0; ks < 4; ++ks)
    w3f[ks] = load_bfrag(wsw + OW3 + ((wv * 4 + ks) * 32 + lane) * 16);

  const int nOut = wv * 16 + lm;      // output feature this lane owns
  const float b1 = W1b[nOut], b2 = W2b[nOut], b3 = W3b[nOut];
  const v8f z = {0.f,0.f,0.f,0.f,0.f,0.f,0.f,0.f};

  // ===================== per-node edge MLP =====================
  for (int nl = 0; nl < 16; ++nl) {
    const int node = nodeBase + nl;
    __syncthreads();                      // protect s_hcat reuse
    {
      // stage h_cat[32][384] bf16: cols 0..127 = h_V broadcast, 128..383 = h_EV
      const int r = tid >> 3, q = tid & 7;
      const float4* src = (const float4*)(hEV + ((size_t)node * 32 + r) * 256) + q * 8;
      unsigned int* dst = (unsigned int*)(s_hcat + r * SC + 128 + q * 32);
#pragma unroll
      for (int j = 0; j < 8; ++j) {
        float4 v = src[j];
        dst[2 * j]     = pack2(v.x, v.y);
        dst[2 * j + 1] = pack2(v.z, v.w);
      }
      const float4* sv = (const float4*)(hV + (size_t)node * 128) + q * 4;
      unsigned int* dv = (unsigned int*)(s_hcat + r * SC + q * 16);
#pragma unroll
      for (int j = 0; j < 4; ++j) {
        float4 v = sv[j];
        dv[2 * j]     = pack2(v.x, v.y);
        dv[2 * j + 1] = pack2(v.z, v.w);
      }
      if (nl + 1 < 16)
        __builtin_prefetch(hEV + ((size_t)(node + 1) * 32) * 256 + tid * 32, 0, 0);
    }
    __syncthreads();

    // GEMM1: [32,384] x W1 -> [32,128], GELU
    v8f a0 = z, a1 = z;
#pragma unroll
    for (int ks = 0; ks < 12; ++ks) {
      v16bf A0 = load_afrag(s_hcat + lm * SC + ks * 32 + 8 * hf);
      a0 = wmma_bf16(A0, w1f[ks], a0);
      v16bf A1 = load_afrag(s_hcat + (16 + lm) * SC + ks * 32 + 8 * hf);
      a1 = wmma_bf16(A1, w1f[ks], a1);
    }
#pragma unroll
    for (int r = 0; r < 8; ++r) {
      int m = r + 8 * hf;
      s_mid[m * SM + nOut]        = f2bf(gelu(a0[r] + b1));
      s_mid[(16 + m) * SM + nOut] = f2bf(gelu(a1[r] + b1));
    }
    __syncthreads();

    // GEMM2: [32,128] x W2 -> [32,128], GELU
    a0 = z; a1 = z;
#pragma unroll
    for (int ks = 0; ks < 4; ++ks) {
      v16bf A0 = load_afrag(s_mid + lm * SM + ks * 32 + 8 * hf);
      a0 = wmma_bf16(A0, w2f[ks], a0);
      v16bf A1 = load_afrag(s_mid + (16 + lm) * SM + ks * 32 + 8 * hf);
      a1 = wmma_bf16(A1, w2f[ks], a1);
    }
#pragma unroll
    for (int r = 0; r < 8; ++r) {
      int m = r + 8 * hf;
      s_mid2[m * SM + nOut]        = f2bf(gelu(a0[r] + b2));
      s_mid2[(16 + m) * SM + nOut] = f2bf(gelu(a1[r] + b2));
    }
    __syncthreads();

    // GEMM3: [32,128] x W3 -> sum over 32 edges / 16
    a0 = z; a1 = z;
#pragma unroll
    for (int ks = 0; ks < 4; ++ks) {
      v16bf A0 = load_afrag(s_mid2 + lm * SM + ks * 32 + 8 * hf);
      a0 = wmma_bf16(A0, w3f[ks], a0);
      v16bf A1 = load_afrag(s_mid2 + (16 + lm) * SM + ks * 32 + 8 * hf);
      a1 = wmma_bf16(A1, w3f[ks], a1);
    }
    float ssum = 0.f;
#pragma unroll
    for (int r = 0; r < 8; ++r) ssum += (a0[r] + b3) + (a1[r] + b3);
    ssum += __shfl_xor(ssum, 16, 32);     // combine row-halves
    if (hf == 0) s_dh[nl * HC + nOut] = ssum * 0.0625f;   // /16
  }
  __syncthreads();

  // ===================== LN1 (wave handles nodes wv, wv+8) ====
#pragma unroll
  for (int i = 0; i < 2; ++i) {
    int nl = wv + i * 8;
    int node = nodeBase + nl;
    float4 xv = *(const float4*)(hV + (size_t)node * 128 + lane * 4);
    float4 dv = *(const float4*)(s_dh + nl * HC + lane * 4);
    float x0 = xv.x + dv.x, x1 = xv.y + dv.y, x2 = xv.z + dv.z, x3 = xv.w + dv.w;
    float s  = x0 + x1 + x2 + x3;
    float sq = x0 * x0 + x1 * x1 + x2 * x2 + x3 * x3;
#pragma unroll
    for (int off = 16; off; off >>= 1) {
      s  += __shfl_xor(s,  off, 32);
      sq += __shfl_xor(sq, off, 32);
    }
    float mean = s * (1.f / 128.f);
    float var  = sq * (1.f / 128.f) - mean * mean;
    float rstd = rsqrtf(var + 1e-5f);
    float4 gv = *(const float4*)(ln1g + lane * 4);
    float4 bv = *(const float4*)(ln1b + lane * 4);
    float y0 = (x0 - mean) * rstd * gv.x + bv.x;
    float y1 = (x1 - mean) * rstd * gv.y + bv.y;
    float y2 = (x2 - mean) * rstd * gv.z + bv.z;
    float y3 = (x3 - mean) * rstd * gv.w + bv.w;
    *(float4*)(s_lnf + nl * HC + lane * 4) = make_float4(y0, y1, y2, y3);
    unsigned int* pb = (unsigned int*)(s_lnb + nl * SM + lane * 4);
    pb[0] = pack2(y0, y1); pb[1] = pack2(y2, y3);
  }
  __syncthreads();

  // ===================== FFN in: [16,128] x Win -> [16,256], GELU
  v16bf wif0[4], wif1[4];
#pragma unroll
  for (int ks = 0; ks < 4; ++ks) {
    wif0[ks] = load_bfrag(wsw + OWIN + (((wv * 2 + 0) * 4 + ks) * 32 + lane) * 16);
    wif1[ks] = load_bfrag(wsw + OWIN + (((wv * 2 + 1) * 4 + ks) * 32 + lane) * 16);
  }
  v8f c0 = z, c1 = z;
#pragma unroll
  for (int ks = 0; ks < 4; ++ks) {
    v16bf A = load_afrag(s_lnb + lm * SM + ks * 32 + 8 * hf);
    c0 = wmma_bf16(A, wif0[ks], c0);
    c1 = wmma_bf16(A, wif1[ks], c1);
  }
  const int nf0 = wv * 32 + lm, nf1 = nf0 + 16;
  const float bi0 = Winb[nf0], bi1 = Winb[nf1];
#pragma unroll
  for (int r = 0; r < 8; ++r) {
    int m = r + 8 * hf;
    s_ff[m * SF + nf0] = f2bf(gelu(c0[r] + bi0));
    s_ff[m * SF + nf1] = f2bf(gelu(c1[r] + bi1));
  }
  __syncthreads();

  // ===================== FFN out: [16,256] x Wout -> [16,128] + residual
  v16bf wof[8];
#pragma unroll
  for (int ks = 0; ks < 8; ++ks)
    wof[ks] = load_bfrag(wsw + OWOUT + ((wv * 8 + ks) * 32 + lane) * 16);
  v8f d = z;
#pragma unroll
  for (int ks = 0; ks < 8; ++ks) {
    v16bf A = load_afrag(s_ff + lm * SF + ks * 32 + 8 * hf);
    d = wmma_bf16(A, wof[ks], d);
  }
  const float bo = Woutb[nOut];
#pragma unroll
  for (int r = 0; r < 8; ++r) {
    int m = r + 8 * hf;
    s_fo[m * HC + nOut] = d[r] + bo + s_lnf[m * HC + nOut];
  }
  __syncthreads();

  // ===================== LN2 -> out ===========================
#pragma unroll
  for (int i = 0; i < 2; ++i) {
    int nl = wv + i * 8;
    int node = nodeBase + nl;
    float4 xv = *(const float4*)(s_fo + nl * HC + lane * 4);
    float x0 = xv.x, x1 = xv.y, x2 = xv.z, x3 = xv.w;
    float s  = x0 + x1 + x2 + x3;
    float sq = x0 * x0 + x1 * x1 + x2 * x2 + x3 * x3;
#pragma unroll
    for (int off = 16; off; off >>= 1) {
      s  += __shfl_xor(s,  off, 32);
      sq += __shfl_xor(sq, off, 32);
    }
    float mean = s * (1.f / 128.f);
    float var  = sq * (1.f / 128.f) - mean * mean;
    float rstd = rsqrtf(var + 1e-5f);
    float4 gv = *(const float4*)(ln2g + lane * 4);
    float4 bv = *(const float4*)(ln2b + lane * 4);
    float4 o;
    o.x = (x0 - mean) * rstd * gv.x + bv.x;
    o.y = (x1 - mean) * rstd * gv.y + bv.y;
    o.z = (x2 - mean) * rstd * gv.z + bv.z;
    o.w = (x3 - mean) * rstd * gv.w + bv.w;
    *(float4*)(out + (size_t)node * 128 + lane * 4) = o;
  }
}

// ---- launch ---------------------------------------------------------------

extern "C" void kernel_launch(void* const* d_in, const int* in_sizes, int n_in,
                              void* d_out, int out_size, void* d_ws, size_t ws_size,
                              hipStream_t stream) {
  const float* hV    = (const float*)d_in[0];
  const float* hEV   = (const float*)d_in[1];
  const float* W1w   = (const float*)d_in[2];
  const float* W1b   = (const float*)d_in[3];
  const float* W2w   = (const float*)d_in[4];
  const float* W2b   = (const float*)d_in[5];
  const float* W3w   = (const float*)d_in[6];
  const float* W3b   = (const float*)d_in[7];
  const float* ln1g  = (const float*)d_in[8];
  const float* ln1b  = (const float*)d_in[9];
  const float* ln2g  = (const float*)d_in[10];
  const float* ln2b  = (const float*)d_in[11];
  const float* Winw  = (const float*)d_in[12];
  const float* Winb  = (const float*)d_in[13];
  const float* Woutw = (const float*)d_in[14];
  const float* Woutb = (const float*)d_in[15];
  unsigned short* ws = (unsigned short*)d_ws;
  float* out = (float*)d_out;

  // pack weights into bf16 WMMA B-fragment order (L2-resident, ~295 KB)
  pack_weights_kernel<<<(49152 + 255) / 256, 256, 0, stream>>>(W1w,  ws + OW1,   384, 128);
  pack_weights_kernel<<<(16384 + 255) / 256, 256, 0, stream>>>(W2w,  ws + OW2,   128, 128);
  pack_weights_kernel<<<(16384 + 255) / 256, 256, 0, stream>>>(W3w,  ws + OW3,   128, 128);
  pack_weights_kernel<<<(32768 + 255) / 256, 256, 0, stream>>>(Winw, ws + OWIN,  128, 256);
  pack_weights_kernel<<<(32768 + 255) / 256, 256, 0, stream>>>(Woutw,ws + OWOUT, 256, 128);

  const int nodes  = in_sizes[0] / 128;   // B*N = 16384
  const int blocks = nodes / 16;          // 16 nodes per block
  declayer_kernel<<<blocks, 256, 0, stream>>>(
      hV, hEV, W1b, W2b, W3b, ln1g, ln1b, ln2g, ln2b, Winb, Woutb, ws, out);
}
